// MultiHead_52097953300793
// MI455X (gfx1250) — matrix operand order repaired and checked by reading, not compiled, not run on hardware
//
#include <hip/hip_runtime.h>
#include <hip/hip_bf16.h>

// MultiHeadAttention forward for MI455X (gfx1250), bf16 WMMA everywhere.
// B=4 T=2048 C=1024 H=16 D=64
#define BB 4
#define TT 2048
#define CC 1024
#define HH 16
#define DDH 64

typedef __attribute__((ext_vector_type(16))) __bf16 bf16x16;
typedef __attribute__((ext_vector_type(8)))  float  f32x8;

union FragU { bf16x16 v; uint4 q[2]; };

__device__ __forceinline__ f32x8 zero8() {
  f32x8 z = {0.f, 0.f, 0.f, 0.f, 0.f, 0.f, 0.f, 0.f};
  return z;
}

__device__ __forceinline__ unsigned short f2bf(float f) {
  unsigned u = __builtin_bit_cast(unsigned, f);
  u += 0x7FFFu + ((u >> 16) & 1u);   // round-to-nearest-even
  return (unsigned short)(u >> 16);
}

// A-fragment (16 x 32 bf16). Per ISA 7.12.2: lane holds row (lane%16);
// elements 0..7 -> K = 8*half + e, elements 8..15 -> K = 16 + 8*half + (e-8).
__device__ __forceinline__ bf16x16 load_fragA(const unsigned short* base, int stride, int lane) {
  const unsigned short* p = base + (size_t)(lane & 15) * stride + ((lane >> 4) << 3);
  FragU f;
  f.q[0] = *(const uint4*)(p);
  f.q[1] = *(const uint4*)(p + 16);
  return f.v;
}

// B-fragment (32 x 16 bf16), loaded from N-major storage (row n = lane%16,
// contiguous in K). Elements 0..7 -> K = 8*half + e, 8..15 -> K = 16 + 8*half + (e-8).
__device__ __forceinline__ bf16x16 load_fragB(const unsigned short* base, int stride, int lane) {
  const unsigned short* p = base + (size_t)(lane & 15) * stride + ((lane >> 4) << 3);
  FragU f;
  f.q[0] = *(const uint4*)(p);
  f.q[1] = *(const uint4*)(p + 16);
  return f.v;
}

__device__ __forceinline__ f32x8 wmma_bf16(bf16x16 a, bf16x16 b, f32x8 c) {
  return __builtin_amdgcn_wmma_f32_16x16x32_bf16(false, a, false, b, (short)0, c, false, false);
}

// ---------------------------------------------------------------- cast kernels
// 8 elements per thread: 2x float4 loads -> one b128 store of 8 bf16.
__global__ void cast_bf16_kernel(const float* __restrict__ in,
                                 unsigned short* __restrict__ out, int n8) {
  int i = blockIdx.x * blockDim.x + threadIdx.x;
  if (i >= n8) return;
  const float4* p = (const float4*)in + (size_t)i * 2;
  float4 f0 = p[0];
  float4 f1 = p[1];
  uint4 pk;
  unsigned short* ps = (unsigned short*)&pk;
  ps[0] = f2bf(f0.x); ps[1] = f2bf(f0.y); ps[2] = f2bf(f0.z); ps[3] = f2bf(f0.w);
  ps[4] = f2bf(f1.x); ps[5] = f2bf(f1.y); ps[6] = f2bf(f1.z); ps[7] = f2bf(f1.w);
  *((uint4*)out + i) = pk;
}

// W[h, k, n] (f32) -> WT[h, n, k] (bf16). Output-indexed: coalesced writes,
// cache-served strided reads.
__global__ void transpose_w_kernel(const float* __restrict__ W,
                                   unsigned short* __restrict__ WT,
                                   int Hn, int Kn, int Nn) {
  int o = blockIdx.x * blockDim.x + threadIdx.x;
  int total = Hn * Kn * Nn;
  if (o >= total) return;
  int kk = o % Kn;
  int n  = (o / Kn) % Nn;
  int hh = o / (Kn * Nn);
  WT[o] = f2bf(W[((size_t)hh * Kn + kk) * Nn + n]);
}

// ------------------------------------------------------- per-head projections
// X: [B*T, C] bf16 ; WT: [H, D, C] bf16 (N-major) ; mode 0: Out=[B,H,T,D],
// mode 1: Out=[B,H,D,T] (transposed, for V).
// 4 waves, each computes a 64x64 tile (acc[4][4]); block covers 256 rows.
// Per K-step: 16 WMMA for 16 b128 loads per lane.
__global__ __launch_bounds__(128) void proj_kernel(const unsigned short* __restrict__ X,
                                                   const unsigned short* __restrict__ WT,
                                                   unsigned short* __restrict__ Out,
                                                   int mode) {
  const int lane = threadIdx.x & 31;
  const int wave = threadIdx.x >> 5;
  const int half = lane >> 4, nn = lane & 15;
  const int h    = blockIdx.y;
  const int row0 = blockIdx.x * 256 + wave * 64;

  f32x8 acc[4][4];
#pragma unroll
  for (int i = 0; i < 4; ++i)
#pragma unroll
    for (int j = 0; j < 4; ++j) acc[i][j] = zero8();

  const unsigned short* wbase = WT + (size_t)h * DDH * CC;
  for (int k0 = 0; k0 < CC; k0 += 32) {
    bf16x16 a[4];
#pragma unroll
    for (int i = 0; i < 4; ++i)
      a[i] = load_fragA(X + (size_t)(row0 + 16 * i) * CC + k0, CC, lane);
#pragma unroll
    for (int j = 0; j < 4; ++j) {
      bf16x16 bf = load_fragB(wbase + (size_t)(j * 16) * CC + k0, CC, lane);
#pragma unroll
      for (int i = 0; i < 4; ++i) acc[i][j] = wmma_bf16(a[i], bf, acc[i][j]);
    }
  }

  const int b  = row0 / TT;
  const int t0 = row0 % TT;
  if (mode == 0) {
#pragma unroll
    for (int i = 0; i < 4; ++i)
#pragma unroll
      for (int j = 0; j < 4; ++j) {
        const int d = j * 16 + nn;
#pragma unroll
        for (int r = 0; r < 8; ++r) {
          const int t = t0 + i * 16 + r + 8 * half;
          Out[((size_t)(b * HH + h) * TT + t) * DDH + d] = f2bf(acc[i][j][r]);
        }
      }
  } else {
#pragma unroll
    for (int i = 0; i < 4; ++i)
#pragma unroll
      for (int j = 0; j < 4; ++j) {
        uint4 pk;
        unsigned short* ps = (unsigned short*)&pk;
#pragma unroll
        for (int r = 0; r < 8; ++r) ps[r] = f2bf(acc[i][j][r]);
        const int d = j * 16 + nn;
        const size_t a = ((size_t)(b * HH + h) * DDH + d) * TT + t0 + i * 16 + 8 * half;
        *(uint4*)(Out + a) = pk;  // C-frag rows are contiguous in t
      }
  }
}

// ------------------------------------------------------------ flash attention
// QH,KH: [B,H,T,D] bf16 ; VT: [B,H,D,T] bf16 ; Out: [B*T, H*D] bf16.
// 4 waves, each owns 16 query rows; block covers 64 rows; streams 64-wide
// key blocks with online softmax. O is accumulated transposed (O^T = V^T P^T)
// so both WMMA operands are natural A/B fragments.
__global__ __launch_bounds__(128) void attn_kernel(const unsigned short* __restrict__ QH,
                                                   const unsigned short* __restrict__ KH,
                                                   const unsigned short* __restrict__ VT,
                                                   unsigned short* __restrict__ Out) {
  __shared__ __align__(16) unsigned short Pbuf[4][16 * 64];  // per-wave P tile [t][s]
  __shared__ float stat[4][16];

  const int lane = threadIdx.x & 31;
  const int wave = threadIdx.x >> 5;
  const int half = lane >> 4, nn = lane & 15;
  const int tb = blockIdx.x;
  const int h  = blockIdx.y;
  const int b  = blockIdx.z;
  const int bh = b * HH + h;
  const unsigned short* Qb = QH + (size_t)bh * TT * DDH;
  const unsigned short* Kb = KH + (size_t)bh * TT * DDH;
  const unsigned short* Vb = VT + (size_t)bh * DDH * TT;
  const int t0w = tb * 64 + wave * 16;

  // Q rows for this wave, K=64 -> 2 A-fragments, held for whole kernel.
  bf16x16 qf0 = load_fragA(Qb + (size_t)t0w * DDH + 0, DDH, lane);
  bf16x16 qf1 = load_fragA(Qb + (size_t)t0w * DDH + 32, DDH, lane);

  f32x8 ot[4];
#pragma unroll
  for (int i = 0; i < 4; ++i) ot[i] = zero8();
  float m8[8], l8[8];
#pragma unroll
  for (int r = 0; r < 8; ++r) { m8[r] = -1e30f; l8[r] = 0.f; }

  const float scale = 0.03125f;  // 1/sqrt(C) = 1/32
  const int nsb = tb + 1;        // uniform across the block -> syncthreads legal
  for (int sb = 0; sb < nsb; ++sb) {
    const int s0 = sb * 64;
    f32x8 sf[4];
#pragma unroll
    for (int j = 0; j < 4; ++j) {
      bf16x16 kf0 = load_fragB(Kb + (size_t)(s0 + j * 16) * DDH + 0, DDH, lane);
      bf16x16 kf1 = load_fragB(Kb + (size_t)(s0 + j * 16) * DDH + 32, DDH, lane);
      f32x8 z = zero8();
      z = wmma_bf16(qf0, kf0, z);
      z = wmma_bf16(qf1, kf1, z);
      sf[j] = z;
    }
    // scale + causal mask (row t = t0w + r + 8*half, col s = s0 + j*16 + nn)
#pragma unroll
    for (int j = 0; j < 4; ++j) {
      const int s = s0 + j * 16 + nn;
#pragma unroll
      for (int r = 0; r < 8; ++r) {
        const int t = t0w + r + 8 * half;
        const float vv = sf[j][r] * scale;
        sf[j][r] = (s <= t) ? vv : -1e30f;
      }
    }
    // online softmax stats; rows live across the 16 lanes of each half
    float alpha[8];
#pragma unroll
    for (int r = 0; r < 8; ++r) {
      float mx = fmaxf(fmaxf(sf[0][r], sf[1][r]), fmaxf(sf[2][r], sf[3][r]));
      mx = fmaxf(mx, __shfl_xor(mx, 1));
      mx = fmaxf(mx, __shfl_xor(mx, 2));
      mx = fmaxf(mx, __shfl_xor(mx, 4));
      mx = fmaxf(mx, __shfl_xor(mx, 8));
      const float mnew = fmaxf(m8[r], mx);
      alpha[r] = __expf(m8[r] - mnew);
      float rs = 0.f;
#pragma unroll
      for (int j = 0; j < 4; ++j) {
        const float p = __expf(sf[j][r] - mnew);
        sf[j][r] = p;
        rs += p;
      }
      rs += __shfl_xor(rs, 1);
      rs += __shfl_xor(rs, 2);
      rs += __shfl_xor(rs, 4);
      rs += __shfl_xor(rs, 8);
      l8[r] = l8[r] * alpha[r] + rs;
      m8[r] = mnew;
    }
    // broadcast per-row alpha (O^T frag indexes rows by lane column nn)
    if (nn == 0) {
#pragma unroll
      for (int r = 0; r < 8; ++r) stat[wave][r + 8 * half] = alpha[r];
    }
    // P -> LDS as bf16 in [t][s] layout (row-major => P^T is a B-fragment)
#pragma unroll
    for (int j = 0; j < 4; ++j)
#pragma unroll
      for (int r = 0; r < 8; ++r)
        Pbuf[wave][(r + 8 * half) * 64 + j * 16 + nn] = f2bf(sf[j][r]);
    __syncthreads();

    const float acol = stat[wave][nn];
#pragma unroll
    for (int i = 0; i < 4; ++i) ot[i] *= acol;

#pragma unroll
    for (int kk = 0; kk < 2; ++kk) {
      bf16x16 pb = load_fragB(&Pbuf[wave][0] + kk * 32, 64, lane);  // ds_load_b128
#pragma unroll
      for (int i = 0; i < 4; ++i) {
        bf16x16 av = load_fragA(Vb + (size_t)(i * 16) * TT + s0 + kk * 32, TT, lane);
        ot[i] = wmma_bf16(av, pb, ot[i]);  // O^T += V^T_tile * P^T_tile
      }
    }
    __syncthreads();
  }

  if (nn == 0) {
#pragma unroll
    for (int r = 0; r < 8; ++r) stat[wave][r + 8 * half] = l8[r];
  }
  __syncthreads();
  const float linv = 1.0f / stat[wave][nn];

  // O^T frag: row = d_local = i*16 + r + 8*half, col = t_local = nn.
  // Packing over r gives 8 contiguous d -> one b128 store per fragment.
#pragma unroll
  for (int i = 0; i < 4; ++i) {
    uint4 pk;
    unsigned short* ps = (unsigned short*)&pk;
#pragma unroll
    for (int r = 0; r < 8; ++r) ps[r] = f2bf(ot[i][r] * linv);
    const size_t a = (size_t)(b * TT + t0w + nn) * (HH * DDH) + h * DDH + i * 16 + 8 * half;
    *(uint4*)(Out + a) = pk;
  }
}

// ----------------------------------------------------------- output projection
// A: [B*T, H*D] bf16 ; WoT: [C, H*D] bf16 (N-major) ; Y: [B*T, C] f32 (+bias)
// 4 waves, each computes 64x64 (acc[4][4]).
__global__ __launch_bounds__(128) void outproj_kernel(const unsigned short* __restrict__ A,
                                                      const unsigned short* __restrict__ WoT,
                                                      const float* __restrict__ bo,
                                                      float* __restrict__ Y) {
  const int lane = threadIdx.x & 31;
  const int wave = threadIdx.x >> 5;
  const int half = lane >> 4, nn = lane & 15;
  const int row0 = blockIdx.x * 256 + wave * 64;
  const int n0   = blockIdx.y * 64;

  f32x8 acc[4][4];
#pragma unroll
  for (int i = 0; i < 4; ++i)
#pragma unroll
    for (int j = 0; j < 4; ++j) acc[i][j] = zero8();

  for (int k0 = 0; k0 < CC; k0 += 32) {   // K = H*D = 1024
    bf16x16 a[4];
#pragma unroll
    for (int i = 0; i < 4; ++i)
      a[i] = load_fragA(A + (size_t)(row0 + 16 * i) * CC + k0, CC, lane);
#pragma unroll
    for (int j = 0; j < 4; ++j) {
      bf16x16 bf = load_fragB(WoT + (size_t)(n0 + j * 16) * CC + k0, CC, lane);
#pragma unroll
      for (int i = 0; i < 4; ++i) acc[i][j] = wmma_bf16(a[i], bf, acc[i][j]);
    }
  }
#pragma unroll
  for (int i = 0; i < 4; ++i)
#pragma unroll
    for (int j = 0; j < 4; ++j) {
      const int col = n0 + j * 16 + nn;
      const float bias = bo[col];
#pragma unroll
      for (int r = 0; r < 8; ++r) {
        const int row = row0 + i * 16 + r + 8 * half;
        Y[(size_t)row * CC + col] = acc[i][j][r] + bias;
      }
    }
}

// ------------------------------------------------------------------- launcher
extern "C" void kernel_launch(void* const* d_in, const int* in_sizes, int n_in,
                              void* d_out, int out_size, void* d_ws, size_t ws_size,
                              hipStream_t stream) {
  (void)in_sizes; (void)n_in; (void)out_size; (void)ws_size;
  const float* q  = (const float*)d_in[0];
  const float* k  = (const float*)d_in[1];
  const float* v  = (const float*)d_in[2];
  const float* Wq = (const float*)d_in[3];
  const float* Wk = (const float*)d_in[4];
  const float* Wv = (const float*)d_in[5];
  const float* Wo = (const float*)d_in[6];
  const float* bo = (const float*)d_in[7];
  float* Y = (float*)d_out;

  // Workspace (bf16), ~69 MB total; xb is reused as the attention output.
  const size_t NE = (size_t)BB * TT * CC;            // 8,388,608 elems
  unsigned short* xb = (unsigned short*)d_ws;        // x bf16 / attn out
  unsigned short* qh = xb + NE;                      // [B,H,T,D]
  unsigned short* kh = qh + NE;                      // [B,H,T,D]
  unsigned short* vt = kh + NE;                      // [B,H,D,T]
  unsigned short* WT = vt + NE;                      // 1M elems, reused per weight

  const int n8 = (int)(NE / 8);
  dim3 cgrid((n8 + 255) / 256);
  dim3 wgrid((HH * CC * DDH + 255) / 256);
  dim3 ogridW((CC * CC + 255) / 256);
  dim3 pgrid(BB * TT / 256, HH), blk(128);
  dim3 agrid(TT / 64, HH, BB);
  dim3 ogrid(BB * TT / 256, CC / 64);

  // Q projection
  transpose_w_kernel<<<wgrid, 256, 0, stream>>>(Wq, WT, HH, CC, DDH);
  cast_bf16_kernel<<<cgrid, 256, 0, stream>>>(q, xb, n8);
  proj_kernel<<<pgrid, blk, 0, stream>>>(xb, WT, qh, 0);
  // K projection
  transpose_w_kernel<<<wgrid, 256, 0, stream>>>(Wk, WT, HH, CC, DDH);
  cast_bf16_kernel<<<cgrid, 256, 0, stream>>>(k, xb, n8);
  proj_kernel<<<pgrid, blk, 0, stream>>>(xb, WT, kh, 0);
  // V projection (stored transposed [B,H,D,T])
  transpose_w_kernel<<<wgrid, 256, 0, stream>>>(Wv, WT, HH, CC, DDH);
  cast_bf16_kernel<<<cgrid, 256, 0, stream>>>(v, xb, n8);
  proj_kernel<<<pgrid, blk, 0, stream>>>(xb, WT, vt, 1);
  // Flash attention -> xb (bf16 [B*T, H*D])
  attn_kernel<<<agrid, blk, 0, stream>>>(qh, kh, vt, xb);
  // Output projection + bias -> f32
  transpose_w_kernel<<<ogridW, 256, 0, stream>>>(Wo, WT, 1, CC, CC);
  outproj_kernel<<<ogrid, blk, 0, stream>>>(xb, WT, bo, Y);
}
